// MultiHeadAttention_34540126994734
// MI455X (gfx1250) — matrix operand order repaired and checked by reading, not compiled, not run on hardware
//
#include <hip/hip_runtime.h>

// ---------------------------------------------------------------------------
// MI455X (gfx1250) multi-head attention forward, bf16 WMMA pipeline with
// async global->LDS staging (inline-asm GLOBAL_LOAD_ASYNC_TO_LDS_B128) and
// double-buffered tiles.  B=2, S=2048, D=1024, H=16, dk=64.
// ---------------------------------------------------------------------------

typedef __attribute__((ext_vector_type(16))) __bf16 v16bf;
typedef __attribute__((ext_vector_type(8)))  __bf16 v8bf;
typedef __attribute__((ext_vector_type(8)))  float  v8f;

static constexpr int BB = 2;
static constexpr int SS = 2048;
static constexpr int DD = 1024;
static constexpr int HH = 16;
static constexpr int DK = 64;
static constexpr int MM = BB * SS;   // 4096 rows for the projections

#define WMMA_BF16(A_, B_, C_) \
  __builtin_amdgcn_wmma_f32_16x16x32_bf16(false, (A_), false, (B_), (short)0, (C_), false, false)

// ---- async global->LDS copy (CDNA5 GLOBAL_LOAD_ASYNC_TO_LDS_B128) ---------
// ISA 15.18.3: LDS destination is the VDST VGPR address plus the offset; tracked by ASYNCcnt.
// Generic pointers to LDS carry the LDS offset in the low 32 bits (ISA 10.2),
// so the VDST operand is just the truncated generic pointer.
#if defined(__HIP_DEVICE_COMPILE__) && defined(__gfx1250__)
#define USE_ASYNC_ASM 1
#endif

static __device__ __forceinline__ void cp_b128(const void* g, void* l) {
#ifdef USE_ASYNC_ASM
  unsigned lds_addr = (unsigned)(uintptr_t)l;
  asm volatile("global_load_async_to_lds_b128 %0, %1, off"
               :
               : "v"(lds_addr), "v"(g)
               : "memory");
#else
  *(v8bf*)l = *(const v8bf*)g;
#endif
}

static __device__ __forceinline__ void async_wait() {
#ifdef USE_ASYNC_ASM
  asm volatile("s_wait_asynccnt 0" ::: "memory");
#endif
}

static __device__ __forceinline__ v8f zero8() {
  v8f z;
#pragma unroll
  for (int i = 0; i < 8; ++i) z[i] = 0.0f;
  return z;
}

// A-matrix 16x32 bf16 fragment (ISA 7.12.2): lane = row (lane&15),
// elements 0..7  = K[half*8    .. half*8+7],
// elements 8..15 = K[16+half*8 .. 16+half*8+7].  tile is row-major, stride ld.
static __device__ __forceinline__ v16bf load_frag_a(const __bf16* tile, int ld, int lane) {
  const int row = lane & 15, half = lane >> 4;
  const __bf16* p = tile + row * ld + half * 8;
  v8bf lo = *(const v8bf*)p;
  v8bf hi = *(const v8bf*)(p + 16);
  v16bf r;
#pragma unroll
  for (int i = 0; i < 8; ++i) { r[i] = lo[i]; r[i + 8] = hi[i]; }
  return r;
}

// B-matrix 32x16 bf16 fragment: lane = column n (lane&15),
// elements 0..7 = K[half*8 .. half*8+7], elements 8..15 = K[16+half*8 .. 16+half*8+7].
// Storage passed here is B^T, i.e. row-major [N][K] with stride ld.
static __device__ __forceinline__ v16bf load_frag_b(const __bf16* tile, int ld, int lane) {
  const int row = lane & 15, half = lane >> 4;
  const __bf16* p = tile + row * ld + half * 8;
  v8bf lo = *(const v8bf*)p;
  v8bf hi = *(const v8bf*)(p + 16);
  v16bf r;
#pragma unroll
  for (int i = 0; i < 8; ++i) { r[i] = lo[i]; r[i + 8] = hi[i]; }
  return r;
}

// ---------------------------------------------------------------------------
// Elementwise conversion kernels
// ---------------------------------------------------------------------------
__global__ void cvt_bf16_kernel(const float* __restrict__ x, __bf16* __restrict__ y, int n) {
  int i = (blockIdx.x * 256 + threadIdx.x) * 4;
  if (i + 3 < n) {
    float4 t = *(const float4*)(x + i);
    y[i + 0] = (__bf16)t.x;
    y[i + 1] = (__bf16)t.y;
    y[i + 2] = (__bf16)t.z;
    y[i + 3] = (__bf16)t.w;
  }
}

// Wt[n][k] = (bf16) W[k][n]   (W stored (in,out), y = x @ W)
__global__ void wtrans_kernel(const float* __restrict__ W, __bf16* __restrict__ Wt) {
  int n = blockIdx.x * 16 + (threadIdx.x & 15);
  int k = blockIdx.y * 16 + (threadIdx.x >> 4);
  Wt[(size_t)n * DD + k] = (__bf16)W[(size_t)k * DD + n];
}

// mask (True = masked key) -> additive bias
__global__ void mask_kernel(const int* __restrict__ m, float* __restrict__ madd, int n) {
  int i = blockIdx.x * 256 + threadIdx.x;
  if (i < n) madd[i] = m[i] ? -1.0e30f : 0.0f;
}

// ---------------------------------------------------------------------------
// GEMM + bias: C[M,N] = A[M,K] (bf16) x Wt^T (Wt is [N][K] bf16) + bias[N]
// 128x128 block tile, K-step 32, 256 threads = 8 waves; wave = 32x64 subtile.
// Double-buffered LDS filled by async global->LDS copies.
// ---------------------------------------------------------------------------
static __device__ __forceinline__ void store_out(float* p, float v)  { *p = v; }
static __device__ __forceinline__ void store_out(__bf16* p, float v) { *p = (__bf16)v; }

template <typename OutT>
__global__ __launch_bounds__(256) void gemm_bias_kernel(
    const __bf16* __restrict__ A, const __bf16* __restrict__ Wt,
    const float* __restrict__ bias, OutT* __restrict__ C,
    int Mdim, int Ndim, int Kdim) {
  __shared__ __bf16 As[2][128][40];   // stride-40 padding: conflict-free frags
  __shared__ __bf16 Bs[2][128][40];

  const int tid  = threadIdx.x;
  const int w    = tid >> 5, lane = tid & 31;
  const int wm   = w & 3;          // 4 waves over M (32 rows each)
  const int wn   = w >> 2;         // 2 waves over N (64 cols each)
  const int m0   = blockIdx.y * 128;
  const int n0   = blockIdx.x * 128;
  const int colL = lane & 15, halfL = lane >> 4;

  auto stage = [&](int buf, int k0) {
#pragma unroll
    for (int i = 0; i < 2; ++i) {
      int c   = tid + 256 * i;        // 512 chunks of 8 bf16 per matrix
      int row = c >> 2;
      int cc  = (c & 3) * 8;
      cp_b128(A  + (size_t)(m0 + row) * Kdim + k0 + cc, &As[buf][row][cc]);
      cp_b128(Wt + (size_t)(n0 + row) * Kdim + k0 + cc, &Bs[buf][row][cc]);
    }
  };

  v8f acc[2][4];
#pragma unroll
  for (int i = 0; i < 2; ++i)
#pragma unroll
    for (int j = 0; j < 4; ++j) acc[i][j] = zero8();

  const int nk = Kdim / 32;
  stage(0, 0);
  for (int ki = 0; ki < nk; ++ki) {
    async_wait();
    __syncthreads();
    const int cur = ki & 1;
    if (ki + 1 < nk) stage(cur ^ 1, (ki + 1) * 32);

    v16bf af[2], bfr[4];
#pragma unroll
    for (int mt = 0; mt < 2; ++mt) af[mt]  = load_frag_a(&As[cur][wm * 32 + mt * 16][0], 40, lane);
#pragma unroll
    for (int nt = 0; nt < 4; ++nt) bfr[nt] = load_frag_b(&Bs[cur][wn * 64 + nt * 16][0], 40, lane);
#pragma unroll
    for (int mt = 0; mt < 2; ++mt)
#pragma unroll
      for (int nt = 0; nt < 4; ++nt) acc[mt][nt] = WMMA_BF16(af[mt], bfr[nt], acc[mt][nt]);
  }

#pragma unroll
  for (int nt = 0; nt < 4; ++nt) {
    int gn   = n0 + wn * 64 + nt * 16 + colL;
    float bv = bias[gn];
#pragma unroll
    for (int mt = 0; mt < 2; ++mt) {
#pragma unroll
      for (int r = 0; r < 8; ++r) {
        int gm = m0 + wm * 32 + mt * 16 + r + 8 * halfL;
        store_out(&C[(size_t)gm * Ndim + gn], acc[mt][nt][r] + bv);
      }
    }
  }
}

// ---------------------------------------------------------------------------
// Flash attention: grid (S/128, H, B), 256 threads = 8 waves x 16 query rows.
// Double-buffered 64-key tiles (async K staging + LDS V transpose);
// online softmax; all matmuls = WMMA.
// ---------------------------------------------------------------------------
__global__ __launch_bounds__(256) void attn_kernel(
    const __bf16* __restrict__ Q, const __bf16* __restrict__ Km,
    const __bf16* __restrict__ Vm, const float* __restrict__ madd,
    __bf16* __restrict__ Ctx) {
  __shared__ __bf16 Ks [2][64][72];     // K tile, row = key, col = dk
  __shared__ __bf16 Vts[2][64][72];     // V^T tile, row = dk, col = key
  __shared__ __bf16 Ps [8][16][72];     // per-wave P staging (C -> A layout)
  __shared__ float  maskS[2][64];

  const int b = blockIdx.z, h = blockIdx.y;
  const int tid = threadIdx.x, w = tid >> 5, lane = tid & 31;
  const int colL = lane & 15, halfL = lane >> 4;
  const int qr = blockIdx.x * 128 + w * 16;     // this wave's 16 query rows

  auto stage_tile = [&](int buf, int kt2) {
    const size_t kbase = ((size_t)(b * SS + kt2 * 64)) * DD + h * DK;
#pragma unroll
    for (int i = 0; i < 2; ++i) {
      int c = tid + 256 * i;                 // 512 chunks of 8
      int row = c >> 3, cc = (c & 7) * 8;
      cp_b128(Km + kbase + (size_t)row * DD + cc, &Ks[buf][row][cc]);
      v8bf vv = *(const v8bf*)(Vm + kbase + (size_t)row * DD + cc);
#pragma unroll
      for (int j = 0; j < 8; ++j) Vts[buf][cc + j][row] = vv[j];  // transpose
    }
    if (tid < 64) maskS[buf][tid] = madd[b * SS + kt2 * 64 + tid];
  };

  const __bf16* Qb = Q + ((size_t)(b * SS + qr)) * DD + h * DK;
  v16bf qf0 = load_frag_a(Qb,      DD, lane);   // dk 0..31
  v16bf qf1 = load_frag_a(Qb + 32, DD, lane);   // dk 32..63

  float mrow[8], lrow[8];
  v8f accO[4];
#pragma unroll
  for (int r = 0; r < 8; ++r) { mrow[r] = -1.0e30f; lrow[r] = 0.0f; }
#pragma unroll
  for (int j = 0; j < 4; ++j) accO[j] = zero8();

  const int nt_tiles = SS / 64;
  stage_tile(0, 0);
  for (int kt = 0; kt < nt_tiles; ++kt) {
    async_wait();
    __syncthreads();
    const int cur = kt & 1;
    if (kt + 1 < nt_tiles) stage_tile(cur ^ 1, kt + 1);

    // scores: 16x64 per wave, scaled + masked
    float pv[4][8];
#pragma unroll
    for (int nt = 0; nt < 4; ++nt) {
      v8f s = zero8();
      s = WMMA_BF16(qf0, load_frag_b(&Ks[cur][nt * 16][0],  72, lane), s);
      s = WMMA_BF16(qf1, load_frag_b(&Ks[cur][nt * 16][32], 72, lane), s);
      float ma = maskS[cur][nt * 16 + colL];
#pragma unroll
      for (int r = 0; r < 8; ++r) pv[nt][r] = s[r] * 0.125f + ma;
    }

    // online softmax: rows live across 16-lane halves -> shfl_xor reductions
    float corr[8];
#pragma unroll
    for (int r = 0; r < 8; ++r) {
      float t = fmaxf(fmaxf(pv[0][r], pv[1][r]), fmaxf(pv[2][r], pv[3][r]));
#pragma unroll
      for (int xm = 1; xm < 16; xm <<= 1) t = fmaxf(t, __shfl_xor(t, xm, 32));
      float mn = fmaxf(mrow[r], t);
      corr[r] = __expf(mrow[r] - mn);
      mrow[r] = mn;
    }
    float tsum[8];
#pragma unroll
    for (int r = 0; r < 8; ++r) tsum[r] = 0.0f;
#pragma unroll
    for (int nt = 0; nt < 4; ++nt)
#pragma unroll
      for (int r = 0; r < 8; ++r) {
        float e = (pv[nt][r] < -1.0e29f) ? 0.0f : __expf(pv[nt][r] - mrow[r]);
        pv[nt][r] = e;
        tsum[r] += e;
      }
#pragma unroll
    for (int r = 0; r < 8; ++r) {
      float t = tsum[r];
#pragma unroll
      for (int xm = 1; xm < 16; xm <<= 1) t += __shfl_xor(t, xm, 32);
      lrow[r] = lrow[r] * corr[r] + t;
    }
#pragma unroll
    for (int j = 0; j < 4; ++j)
#pragma unroll
      for (int r = 0; r < 8; ++r) accO[j][r] *= corr[r];

    // stage P (C-layout) into LDS so it can be re-read in A-fragment layout
#pragma unroll
    for (int nt = 0; nt < 4; ++nt)
#pragma unroll
      for (int r = 0; r < 8; ++r)
        Ps[w][r + 8 * halfL][nt * 16 + colL] = (__bf16)pv[nt][r];
    // (DS ops from one wave execute in order: store->load is safe)

#pragma unroll
    for (int kc = 0; kc < 64; kc += 32) {
      v16bf pf = load_frag_a(&Ps[w][0][kc], 72, lane);
#pragma unroll
      for (int j = 0; j < 4; ++j)
        accO[j] = WMMA_BF16(pf, load_frag_b(&Vts[cur][j * 16][kc], 72, lane), accO[j]);
    }
  }

  // normalize + write context (fully-masked rows -> 0, matching nan_to_num)
#pragma unroll
  for (int r = 0; r < 8; ++r) {
    float inv = lrow[r] > 0.0f ? 1.0f / lrow[r] : 0.0f;
    int row = qr + r + 8 * halfL;
    size_t base = ((size_t)(b * SS + row)) * DD + h * DK;
#pragma unroll
    for (int j = 0; j < 4; ++j)
      Ctx[base + j * 16 + colL] = (__bf16)(accO[j][r] * inv);
  }
}

// ---------------------------------------------------------------------------
// Host launcher
// ---------------------------------------------------------------------------
extern "C" void kernel_launch(void* const* d_in, const int* in_sizes, int n_in,
                              void* d_out, int out_size, void* d_ws, size_t ws_size,
                              hipStream_t stream) {
  const float* q    = (const float*)d_in[0];
  const float* k    = (const float*)d_in[1];
  const float* v    = (const float*)d_in[2];
  const int*   mask = (const int*)  d_in[3];
  const float* Wq = (const float*)d_in[4];  const float* bq = (const float*)d_in[5];
  const float* Wk = (const float*)d_in[6];  const float* bk = (const float*)d_in[7];
  const float* Wv = (const float*)d_in[8];  const float* bv = (const float*)d_in[9];
  const float* Wo = (const float*)d_in[10]; const float* bo = (const float*)d_in[11];

  char* p = (char*)d_ws;
  auto take = [&](size_t bytes) {
    char* r = p;
    p += (bytes + 255) & ~size_t(255);
    return r;
  };
  const size_t xbytes = (size_t)MM * DD * sizeof(__bf16);   // 8 MB
  const size_t wbytes = (size_t)DD * DD * sizeof(__bf16);   // 2 MB
  __bf16* Xq   = (__bf16*)take(xbytes);
  __bf16* Xk   = (__bf16*)take(xbytes);
  __bf16* Xv   = (__bf16*)take(xbytes);
  __bf16* Wtq  = (__bf16*)take(wbytes);
  __bf16* Wtk  = (__bf16*)take(wbytes);
  __bf16* Wtv  = (__bf16*)take(wbytes);
  __bf16* Wto  = (__bf16*)take(wbytes);
  __bf16* Qp   = (__bf16*)take(xbytes);
  __bf16* Kp   = (__bf16*)take(xbytes);
  __bf16* Vp   = (__bf16*)take(xbytes);
  __bf16* Ctxb = (__bf16*)take(xbytes);
  float*  madd = (float*) take((size_t)BB * SS * sizeof(float));

  const dim3 blk(256);
  const int ncvt = MM * DD;                       // 4M elements each
  cvt_bf16_kernel<<<ncvt / 1024, blk, 0, stream>>>(q, Xq, ncvt);
  cvt_bf16_kernel<<<ncvt / 1024, blk, 0, stream>>>(k, Xk, ncvt);
  cvt_bf16_kernel<<<ncvt / 1024, blk, 0, stream>>>(v, Xv, ncvt);
  wtrans_kernel<<<dim3(64, 64), blk, 0, stream>>>(Wq, Wtq);
  wtrans_kernel<<<dim3(64, 64), blk, 0, stream>>>(Wk, Wtk);
  wtrans_kernel<<<dim3(64, 64), blk, 0, stream>>>(Wv, Wtv);
  wtrans_kernel<<<dim3(64, 64), blk, 0, stream>>>(Wo, Wto);
  mask_kernel<<<(BB * SS + 255) / 256, blk, 0, stream>>>(mask, madd, BB * SS);

  const dim3 ggrid(DD / 128, MM / 128);           // (8, 32)
  gemm_bias_kernel<__bf16><<<ggrid, blk, 0, stream>>>(Xq, Wtq, bq, Qp, MM, DD, DD);
  gemm_bias_kernel<__bf16><<<ggrid, blk, 0, stream>>>(Xk, Wtk, bk, Kp, MM, DD, DD);
  gemm_bias_kernel<__bf16><<<ggrid, blk, 0, stream>>>(Xv, Wtv, bv, Vp, MM, DD, DD);

  attn_kernel<<<dim3(SS / 128, HH, BB), blk, 0, stream>>>(Qp, Kp, Vp, madd, Ctxb);

  gemm_bias_kernel<float><<<ggrid, blk, 0, stream>>>(Ctxb, Wto, bo, (float*)d_out, MM, DD, DD);
}
